// CausalSelfAttention_17179869184712
// MI455X (gfx1250) — compile-verified, matrix operand order checked by reading, not run on hardware
//
#include <hip/hip_runtime.h>

// ---------------------------------------------------------------------------
// CausalSelfAttention (GQA) on MI455X / gfx1250 — bf16 WMMA pipeline
//   B=4, T=2048, C=1024, H=16, KV=4 (group size 4), hd=64
// Pipeline:
//   cvt x -> bf16 ; pack weights -> B-fragment tile layout (bf16)
//   Q/K/V projections (WMMA GEMM, fused bias (+qk_gain for Q); V written
//   transposed per batch [b][dim][t] for contiguous attention fragments)
//   flash attention (WMMA QK^T / PV, online softmax, causal chunk skip,
//   mask ops only on the single diagonal boundary chunk)
//   output projection (WMMA GEMM, fused bias, f32 out)
// ---------------------------------------------------------------------------

typedef __bf16 v16bf __attribute__((ext_vector_type(16)));
typedef float  v8f   __attribute__((ext_vector_type(8)));

union Frag {
    v16bf          bf;
    unsigned short us[16];
    unsigned int   u32[8];
};

constexpr int Bsz  = 4;
constexpr int Tlen = 2048;
constexpr int Cdim = 1024;
constexpr int NH   = 16;
constexpr int HD   = 64;
constexpr int KVD  = 256;              // N_KV * hd
constexpr int Mrows = Bsz * Tlen;      // 8192

__device__ __forceinline__ unsigned short f2bf(float f) {
    unsigned u = __float_as_uint(f);
    unsigned r = u + 0x7FFFu + ((u >> 16) & 1u);   // round-to-nearest-even
    return (unsigned short)(r >> 16);
}

__device__ __forceinline__ v8f vzero() {
    v8f v = {0.f, 0.f, 0.f, 0.f, 0.f, 0.f, 0.f, 0.f};
    return v;
}

// ISA 16-bit A/B fragment k-index map (16x32): VGPR j<4 -> k = 8*half + 2j,
// VGPR j>=4 -> k = 16 + 8*half + 2(j-4); each VGPR packs (k, k+1).
// For fixed half this is two contiguous 16B runs -> paired global_load_b128.
__device__ __forceinline__ int kmap(int j, int half) {
    return (j < 4) ? (half * 8 + 2 * j) : (16 + half * 8 + 2 * (j - 4));
}

__device__ __forceinline__ v8f wmma_bf16(v16bf a, v16bf b, v8f c) {
    return __builtin_amdgcn_wmma_f32_16x16x32_bf16(
        /*neg_a=*/false, a, /*neg_b=*/false, b,
        /*c_mod=*/(short)0, c, /*reuse_a=*/false, /*reuse_b=*/false);
}

// ---------------------------------------------------------------------------
// Conversions / weight packing
// ---------------------------------------------------------------------------
__global__ void cvt_f32_to_bf16(const float* __restrict__ in,
                                unsigned short* __restrict__ out, int n) {
    int i = blockIdx.x * blockDim.x + threadIdx.x;
    if (i < n) out[i] = f2bf(in[i]);
}

// Pack torch Linear weight W [N,K] f32 into WMMA B-fragment tiles (bf16):
// tile (nt,kt) = 32 lanes x 8 u32 pairs; pair (lane,j) holds
//   ( k = kt*32 + kmap(j, lane>>4), n = nt*16 + (lane&15) )  and k+1.
// A wave's B-fragment load is then 32 contiguous bytes per lane.
__global__ void pack_weightB_bf16(const float* __restrict__ W,
                                  unsigned int* __restrict__ out,
                                  int N, int K) {
    const int total = (N / 16) * (K / 32) * 256;
    int i = blockIdx.x * blockDim.x + threadIdx.x;
    if (i >= total) return;
    const int j    = i & 7;
    const int lane = (i >> 3) & 31;
    const int kt   = (i >> 8) % (K / 32);
    const int nt   = (i >> 8) / (K / 32);
    const int n = nt * 16 + (lane & 15);
    const int k = kt * 32 + kmap(j, lane >> 4);
    const unsigned short lo = f2bf(W[(size_t)n * K + k]);
    const unsigned short hi = f2bf(W[(size_t)n * K + k + 1]);
    out[i] = (unsigned)lo | ((unsigned)hi << 16);
}

// ---------------------------------------------------------------------------
// WMMA GEMM:  out[M,N] = A[M,K](bf16) * W^T (packed B tiles) + bias
// Block: 256 threads = 8 waves (4 in M x 2 in N); wave tile 32x64 (8 WMMAs/k).
// OM: 0 = bf16 row-major, 1 = f32 row-major, 2 = bf16 transposed [b][col][t]
// ---------------------------------------------------------------------------
template <int OM>
__global__ __launch_bounds__(256)
void gemm_bf16_wmma(const unsigned short* __restrict__ A,
                    const unsigned int* __restrict__ Bpk,
                    const float* __restrict__ bias,
                    const float* __restrict__ gains,   // per (col>>6) gain or null
                    void* __restrict__ outp,
                    int M, int N, int K) {
    const int lane = threadIdx.x & 31;
    const int wave = threadIdx.x >> 5;
    const int half = lane >> 4;
    const int mn   = lane & 15;
    const int row0 = blockIdx.x * 128 + (wave & 3) * 32;
    const int col0 = blockIdx.y * 128 + (wave >> 2) * 64;
    const int ktiles = K >> 5;
    const int nt0    = col0 >> 4;

    v8f acc[2][4];
#pragma unroll
    for (int m = 0; m < 2; ++m)
#pragma unroll
        for (int c = 0; c < 4; ++c) acc[m][c] = vzero();

    const unsigned short* arow0 = A + (size_t)(row0 + mn) * K;
    const unsigned short* arow1 = A + (size_t)(row0 + 16 + mn) * K;

    for (int k0 = 0; k0 < K; k0 += 32) {
        Frag a0, a1;
#pragma unroll
        for (int j = 0; j < 8; ++j) {
            const int kk = kmap(j, half);
            a0.u32[j] = *(const unsigned int*)(arow0 + k0 + kk);
            a1.u32[j] = *(const unsigned int*)(arow1 + k0 + kk);
        }
#pragma unroll
        for (int c = 0; c < 4; ++c) {
            Frag b;
            const unsigned int* bp =
                Bpk + ((size_t)(nt0 + c) * ktiles + (k0 >> 5)) * 256 + lane * 8;
#pragma unroll
            for (int j = 0; j < 8; ++j) b.u32[j] = bp[j];
            acc[0][c] = wmma_bf16(a0.bf, b.bf, acc[0][c]);
            acc[1][c] = wmma_bf16(a1.bf, b.bf, acc[1][c]);
        }
    }

#pragma unroll
    for (int m = 0; m < 2; ++m) {
#pragma unroll
        for (int c = 0; c < 4; ++c) {
            const int col = col0 + c * 16 + mn;
            const float bv = bias[col];
            const float g  = gains ? gains[col >> 6] : 1.0f;
#pragma unroll
            for (int r = 0; r < 8; ++r) {
                const int row = row0 + m * 16 + r + 8 * half;  // C-frag M map
                const float v = (acc[m][c][r] + bv) * g;
                if (OM == 0) {
                    ((unsigned short*)outp)[(size_t)row * N + col] = f2bf(v);
                } else if (OM == 1) {
                    ((float*)outp)[(size_t)row * N + col] = v;
                } else {  // transposed per batch: [b][col][t]
                    const int brow = row / Tlen;
                    const int t    = row - brow * Tlen;
                    ((unsigned short*)outp)[((size_t)brow * N + col) * Tlen + t] =
                        f2bf(v);
                }
            }
        }
    }
}

// ---------------------------------------------------------------------------
// Flash attention chunk: 32 keys against one 16-row query tile.
// MASKED=true only for the single diagonal boundary chunk.
// ---------------------------------------------------------------------------
template <bool MASKED>
__device__ __forceinline__ void attn_chunk(
    int j0, int q0, int half, int mn,
    const Frag* qa, const unsigned short* Kbase, const unsigned short* VtBase,
    unsigned short* pb, v8f* y, float* mst, float* lst) {
    // ---- S = Q * K^T (two 16x16 tiles over 32 keys) ----
    v8f S[2];
    S[0] = vzero();
    S[1] = vzero();
#pragma unroll
    for (int t = 0; t < 2; ++t) {
        const unsigned short* krow = Kbase + (size_t)(j0 + t * 16 + mn) * KVD;
#pragma unroll
        for (int s = 0; s < 2; ++s) {
            Frag kb;
#pragma unroll
            for (int j = 0; j < 8; ++j)
                kb.u32[j] = *(const unsigned int*)(krow + s * 32 + kmap(j, half));
            S[t] = wmma_bf16(qa[s].bf, kb.bf, S[t]);
        }
    }
    if (j0 + 32 < Tlen)   // prefetch next K tile while softmax runs
        __builtin_prefetch(Kbase + (size_t)(j0 + 32 + mn) * KVD, 0, 1);

    // ---- scale, (boundary-only) causal mask, online softmax ----
#pragma unroll
    for (int r = 0; r < 8; ++r) {
        float s0 = S[0][r] * 0.125f;   // 1/sqrt(64)
        float s1 = S[1][r] * 0.125f;
        if (MASKED) {
            const int qr = q0 + r + 8 * half;
            if (j0 + mn > qr)      s0 = -1e30f;
            if (j0 + 16 + mn > qr) s1 = -1e30f;
        }
        float mx = fmaxf(s0, s1);
#pragma unroll
        for (int d = 1; d < 16; d <<= 1) mx = fmaxf(mx, __shfl_xor(mx, d, 32));
        const float mnew = fmaxf(mst[r], mx);
        const float p0 = __expf(s0 - mnew);
        const float p1 = __expf(s1 - mnew);
        float rs = p0 + p1;
#pragma unroll
        for (int d = 1; d < 16; d <<= 1) rs += __shfl_xor(rs, d, 32);
        const float corr = __expf(mst[r] - mnew);
        lst[r] = lst[r] * corr + rs;
        mst[r] = mnew;
#pragma unroll
        for (int c = 0; c < 4; ++c) y[c][r] *= corr;
        const int Mr = r + 8 * half;
        pb[Mr * 34 + mn]      = f2bf(p0);
        pb[Mr * 34 + 16 + mn] = f2bf(p1);
    }

    // C-fragment -> A-fragment re-layout of P through LDS
    asm volatile("s_wait_dscnt 0" ::: "memory");
    Frag pa;
#pragma unroll
    for (int j = 0; j < 8; ++j)
        pa.u32[j] = *(const unsigned int*)&pb[mn * 34 + kmap(j, half)];

    // ---- y += P * V  (V transposed [dim][t] -> contiguous-k fragments) ----
#pragma unroll
    for (int c = 0; c < 4; ++c) {
        Frag vb;
        const unsigned short* vp = VtBase + (size_t)(c * 16 + mn) * Tlen + j0;
#pragma unroll
        for (int j = 0; j < 8; ++j)
            vb.u32[j] = *(const unsigned int*)(vp + kmap(j, half));
        y[c] = wmma_bf16(pa.bf, vb.bf, y[c]);
    }
}

// ---------------------------------------------------------------------------
// Flash attention (causal, GQA): 1 wave = 16 query rows of one head.
// Q[BT,C] bf16 (gain pre-applied), K[BT,KVD] bf16, Vt[b][KVD][T] bf16.
// ---------------------------------------------------------------------------
__global__ __launch_bounds__(256)
void flash_attn_wmma(const unsigned short* __restrict__ Qb,
                     const unsigned short* __restrict__ Kb,
                     const unsigned short* __restrict__ Vt,
                     unsigned short* __restrict__ Yb) {
    __shared__ __attribute__((aligned(16))) unsigned short pbuf[8][16 * 34];

    const int lane = threadIdx.x & 31;
    const int wave = threadIdx.x >> 5;
    const int half = lane >> 4;
    const int mn   = lane & 15;
    const int h    = blockIdx.y;            // query head
    const int b    = blockIdx.z;
    const int kv   = h >> 2;                // shared kv head (G = 4)
    const int q0   = (blockIdx.x * 8 + wave) * 16;

    // Q fragments: 16 rows x 64 dims -> two 16x32 A-fragments (loaded once)
    Frag qa[2];
    {
        const unsigned short* qrow =
            Qb + (size_t)(b * Tlen + q0 + mn) * Cdim + h * HD;
#pragma unroll
        for (int s = 0; s < 2; ++s)
#pragma unroll
            for (int j = 0; j < 8; ++j)
                qa[s].u32[j] = *(const unsigned int*)(qrow + s * 32 + kmap(j, half));
    }

    v8f y[4];
#pragma unroll
    for (int c = 0; c < 4; ++c) y[c] = vzero();
    float mst[8], lst[8];
#pragma unroll
    for (int r = 0; r < 8; ++r) { mst[r] = -1e30f; lst[r] = 0.f; }

    const unsigned short* Kbase  = Kb + (size_t)b * Tlen * KVD + kv * HD;
    const unsigned short* VtBase = Vt + ((size_t)b * KVD + kv * HD) * Tlen;
    unsigned short* pb = pbuf[wave];

    int j0 = 0;
    // mask-free interior chunks: all 32 keys <= every query row in the tile
    for (; j0 + 31 <= q0; j0 += 32)
        attn_chunk<false>(j0, q0, half, mn, qa, Kbase, VtBase, pb, y, mst, lst);
    // exactly one diagonal boundary chunk needs the causal mask
    for (; j0 < q0 + 16; j0 += 32)
        attn_chunk<true>(j0, q0, half, mn, qa, Kbase, VtBase, pb, y, mst, lst);

    // ---- normalize and store bf16 ----
#pragma unroll
    for (int r = 0; r < 8; ++r) {
        const float inv = 1.0f / lst[r];
        const int row = b * Tlen + q0 + r + 8 * half;
#pragma unroll
        for (int c = 0; c < 4; ++c)
            Yb[(size_t)row * Cdim + h * HD + c * 16 + mn] = f2bf(y[c][r] * inv);
    }
}

// ---------------------------------------------------------------------------
// Host-side orchestration
// ---------------------------------------------------------------------------
extern "C" void kernel_launch(void* const* d_in, const int* in_sizes, int n_in,
                              void* d_out, int out_size, void* d_ws, size_t ws_size,
                              hipStream_t stream) {
    const float* x    = (const float*)d_in[0];
    const float* Wq   = (const float*)d_in[1];
    const float* bq   = (const float*)d_in[2];
    const float* Wk   = (const float*)d_in[3];
    const float* bk   = (const float*)d_in[4];
    const float* Wv   = (const float*)d_in[5];
    const float* bv   = (const float*)d_in[6];
    const float* Wo   = (const float*)d_in[7];
    const float* bo   = (const float*)d_in[8];
    const float* gain = (const float*)d_in[9];

    char*  ws  = (char*)d_ws;
    size_t off = 0;
    auto alloc = [&](size_t bytes) {
        void* p = ws + off;
        off += (bytes + 255) & ~(size_t)255;
        return p;
    };

    unsigned short* xbf = (unsigned short*)alloc((size_t)Mrows * Cdim * 2);
    unsigned int*   Wqp = (unsigned int*)alloc((size_t)Cdim * Cdim * 2);
    unsigned int*   Wkp = (unsigned int*)alloc((size_t)KVD * Cdim * 2);
    unsigned int*   Wvp = (unsigned int*)alloc((size_t)KVD * Cdim * 2);
    unsigned int*   Wop = (unsigned int*)alloc((size_t)Cdim * Cdim * 2);
    unsigned short* Qbf = (unsigned short*)alloc((size_t)Mrows * Cdim * 2);
    unsigned short* Kbf = (unsigned short*)alloc((size_t)Mrows * KVD * 2);
    unsigned short* Vt  = (unsigned short*)alloc((size_t)Mrows * KVD * 2);
    unsigned short* Ybf = (unsigned short*)alloc((size_t)Mrows * Cdim * 2);

    const int nx = Mrows * Cdim;
    cvt_f32_to_bf16<<<dim3((nx + 255) / 256), dim3(256), 0, stream>>>(x, xbf, nx);

    const int npq = (Cdim / 16) * (Cdim / 32) * 256;   // pairs in packed Wq/Wo
    const int npk = (KVD / 16) * (Cdim / 32) * 256;    // pairs in packed Wk/Wv
    pack_weightB_bf16<<<dim3((npq + 255) / 256), dim3(256), 0, stream>>>(Wq, Wqp, Cdim, Cdim);
    pack_weightB_bf16<<<dim3((npk + 255) / 256), dim3(256), 0, stream>>>(Wk, Wkp, KVD, Cdim);
    pack_weightB_bf16<<<dim3((npk + 255) / 256), dim3(256), 0, stream>>>(Wv, Wvp, KVD, Cdim);
    pack_weightB_bf16<<<dim3((npq + 255) / 256), dim3(256), 0, stream>>>(Wo, Wop, Cdim, Cdim);

    // Q projection: fuse bias + per-head qk_gain (head = col>>6)
    gemm_bf16_wmma<0><<<dim3(Mrows / 128, Cdim / 128), dim3(256), 0, stream>>>(
        xbf, Wqp, bq, gain, Qbf, Mrows, Cdim, Cdim);
    // K projection (row-major), V projection (transposed per batch)
    gemm_bf16_wmma<0><<<dim3(Mrows / 128, KVD / 128), dim3(256), 0, stream>>>(
        xbf, Wkp, bk, nullptr, Kbf, Mrows, KVD, Cdim);
    gemm_bf16_wmma<2><<<dim3(Mrows / 128, KVD / 128), dim3(256), 0, stream>>>(
        xbf, Wvp, bv, nullptr, Vt, Mrows, KVD, Cdim);

    // Flash attention: 8 q-tiles per block, per head, per batch
    flash_attn_wmma<<<dim3(Tlen / 128, NH, Bsz), dim3(256), 0, stream>>>(Qbf, Kbf, Vt, Ybf);

    // Output projection -> f32
    gemm_bf16_wmma<1><<<dim3(Mrows / 128, Cdim / 128), dim3(256), 0, stream>>>(
        Ybf, Wop, bo, (const float*)nullptr, d_out, Mrows, Cdim, Cdim);
}
